// Cls_HEAD_3985729650694
// MI455X (gfx1250) — matrix operand order repaired and checked by reading, not compiled, not run on hardware
//
#include <hip/hip_runtime.h>
#include <cmath>

// ---------------------------------------------------------------------------
// MoE classifier head for MI455X (gfx1250), wave32 + V_WMMA_F32_16X16X32_BF16.
// fp32 operands are split into bf16 hi/lo; each tile MAC uses 3 WMMAs
// (hi*hi + hi*lo + lo*hi) accumulated in f32 -> ~fp32-accurate, bf16-rate.
// Global loads are float4 (global_load_b128), software-pipelined through a
// double-buffered LDS stage (one barrier per K-step).
// ---------------------------------------------------------------------------

typedef __attribute__((ext_vector_type(16))) __bf16 v16bf;
typedef __attribute__((ext_vector_type(2)))  __bf16 bf16x2;
typedef __attribute__((ext_vector_type(8)))  float  v8f;

#define E_  8
#define B_  4096
#define L_  512
#define H1_ 1024
#define H2_ 512
#define C_  40

// ---------------- routing: histogram -> scan -> compaction -----------------
// ws header: hdr[0..7]=counts, hdr[8..15]=offsets, hdr[16..23]=fill cursors.

__global__ void k_zero_hdr(int* hdr) {
  if (threadIdx.x < 32) hdr[threadIdx.x] = 0;
}

__global__ void k_count(const int* __restrict__ label, int* __restrict__ hdr) {
  int b = blockIdx.x * blockDim.x + threadIdx.x;
  if (b < B_) atomicAdd(&hdr[label[b]], 1);
}

__global__ void k_scan(int* hdr) {
  if (threadIdx.x == 0) {
    int s = 0;
    for (int e = 0; e < E_; ++e) { hdr[8 + e] = s; hdr[16 + e] = s; s += hdr[e]; }
  }
}

__global__ void k_fill(const int* __restrict__ label, int* __restrict__ hdr,
                       int* __restrict__ rows) {
  int b = blockIdx.x * blockDim.x + threadIdx.x;
  if (b < B_) {
    int e = label[b];
    int p = atomicAdd(&hdr[16 + e], 1);
    rows[p] = b;  // compact row p (within expert segment) -> sample b
  }
}

// ------------------------------ helpers ------------------------------------

__device__ __forceinline__ v8f wmma_bf16(v16bf a, v16bf b, v8f c) {
  // 8 args: (neg_a, A, neg_b, B, c_mod, C, reuse_a, reuse_b)
  return __builtin_amdgcn_wmma_f32_16x16x32_bf16(false, a, false, b,
                                                 (short)0, c, false, false);
}

__device__ __forceinline__ unsigned pack2(__bf16 a, __bf16 b) {
  bf16x2 v = {a, b};
  return __builtin_bit_cast(unsigned, v);
}

// --------------------------- fused GEMM layer ------------------------------
// Block = 256 threads (8 wave32), tile = 64(M) x 64(N), K stepped by 32.
// Wave w: M-band = (w&3)*16, N-halves at ((w>>2)*32) and +16 -> 2 acc tiles.
// LDS tiles: 17-dword row pitch (17 coprime 64 banks -> conflict-free),
// double-buffered for a one-barrier software pipeline.
// IN_MODE : 0 = gather concat(x_p,x_s) rows via row map, 1 = compact scratch
// OUT_MODE: 0 = compact scratch [rows, NDIM],            1 = scatter to d_out

template<int KDIM, int NDIM, bool DO_ELU, int IN_MODE, int OUT_MODE>
__global__ __launch_bounds__(256) void moe_mlp_layer(
    const float* __restrict__ inA0,   // IN_MODE==0: x_p ; IN_MODE==1: h
    const float* __restrict__ inA1,   // IN_MODE==0: x_s ; else unused
    const float* __restrict__ W,      // [E, KDIM, NDIM]
    const float* __restrict__ bias,   // [E, NDIM]
    const int*   __restrict__ hdr,
    const int*   __restrict__ rows,
    float* __restrict__ outp)
{
  const int e   = blockIdx.z;
  const int cnt = hdr[e];
  const int m0  = blockIdx.x * 64;
  if (m0 >= cnt) return;               // uniform early-exit (before barriers)
  const int off = hdr[8 + e];
  const int n0  = blockIdx.y * 64;
  const float* We = W    + (size_t)e * KDIM * NDIM;
  const float* be = bias + (size_t)e * NDIM;

  __shared__ unsigned Ah[2][64][17];   // A hi, [m][k-pairs], padded pitch
  __shared__ unsigned Al[2][64][17];   // A lo
  __shared__ unsigned Bh[2][64][17];   // B hi, transposed [n][k-pairs]
  __shared__ unsigned Bl[2][64][17];   // B lo

  const int t    = threadIdx.x;
  const int lane = t & 31;
  const int wave = t >> 5;
  const int mrow = wave & 3;           // 16-row band within tile
  const int nc0  = (wave >> 2) * 32;   // 0 or 32
  const int g    = lane >> 4;          // lane half
  const int l15  = lane & 15;

  // ---- per-thread load assignments (fixed for whole K loop) ----
  const int rA = t >> 3;               // A row 0..31 (and rA+32)
  const int kA = (t & 7) * 4;          // A k offset, float4 granular
  const int kB = t >> 4;               // B k row 0..15 (and kB+16)
  const int nB = (t & 15) * 4;         // B n offset, float4 granular

  const bool vA0 = (m0 + rA)      < cnt;
  const bool vA1 = (m0 + rA + 32) < cnt;
  const bool vB  = (n0 + nB)      < NDIM;   // NDIM % 4 == 0 -> vec4 granular

  // hoisted gather indices / base pointers (loop-invariant)
  int bRow0 = 0, bRow1 = 0;
  const float* aptr0 = nullptr;
  const float* aptr1 = nullptr;
  if (IN_MODE == 0) {
    if (vA0) bRow0 = rows[off + m0 + rA];
    if (vA1) bRow1 = rows[off + m0 + rA + 32];
  } else {
    aptr0 = inA0 + (size_t)(off + m0 + rA) * KDIM + kA;
    aptr1 = aptr0 + (size_t)32 * KDIM;
  }
  const float* bptr0 = We + (size_t)kB * NDIM + (n0 + nB);
  const float* bptr1 = bptr0 + (size_t)16 * NDIM;

  const float4 f4z = {0.f, 0.f, 0.f, 0.f};

  auto loadA = [&](int k0, float4& x0, float4& x1) {
    x0 = f4z; x1 = f4z;
    if (IN_MODE == 0) {
      int k = k0 + kA;
      if (vA0) {
        const float* p = (k < L_) ? inA0 + (size_t)bRow0 * L_ + k
                                  : inA1 + (size_t)bRow0 * L_ + (k - L_);
        x0 = *(const float4*)p;
      }
      if (vA1) {
        const float* p = (k < L_) ? inA0 + (size_t)bRow1 * L_ + k
                                  : inA1 + (size_t)bRow1 * L_ + (k - L_);
        x1 = *(const float4*)p;
      }
    } else {
      if (vA0) x0 = *(const float4*)(aptr0 + k0);
      if (vA1) x1 = *(const float4*)(aptr1 + k0);
    }
  };
  auto loadB = [&](int k0, float4& y0, float4& y1) {
    y0 = f4z; y1 = f4z;
    if (vB) {
      y0 = *(const float4*)(bptr0 + (size_t)k0 * NDIM);
      y1 = *(const float4*)(bptr1 + (size_t)k0 * NDIM);
    }
  };

  auto stageA = [&](int buf, int r, float4 v) {
    __bf16 h0 = (__bf16)v.x, h1 = (__bf16)v.y, h2 = (__bf16)v.z, h3 = (__bf16)v.w;
    __bf16 l0 = (__bf16)(v.x - (float)h0), l1 = (__bf16)(v.y - (float)h1);
    __bf16 l2 = (__bf16)(v.z - (float)h2), l3 = (__bf16)(v.w - (float)h3);
    int c = kA >> 1;
    Ah[buf][r][c]     = pack2(h0, h1);
    Ah[buf][r][c + 1] = pack2(h2, h3);
    Al[buf][r][c]     = pack2(l0, l1);
    Al[buf][r][c + 1] = pack2(l2, l3);
  };
  auto stageB = [&](int buf, int k, float4 v) {
    // transpose: 4 consecutive n -> 4 LDS rows, column k
    __bf16 h0 = (__bf16)v.x, h1 = (__bf16)v.y, h2 = (__bf16)v.z, h3 = (__bf16)v.w;
    __bf16 l0 = (__bf16)(v.x - (float)h0), l1 = (__bf16)(v.y - (float)h1);
    __bf16 l2 = (__bf16)(v.z - (float)h2), l3 = (__bf16)(v.w - (float)h3);
    ((__bf16*)Bh[buf][nB + 0])[k] = h0; ((__bf16*)Bl[buf][nB + 0])[k] = l0;
    ((__bf16*)Bh[buf][nB + 1])[k] = h1; ((__bf16*)Bl[buf][nB + 1])[k] = l1;
    ((__bf16*)Bh[buf][nB + 2])[k] = h2; ((__bf16*)Bl[buf][nB + 2])[k] = l2;
    ((__bf16*)Bh[buf][nB + 3])[k] = h3; ((__bf16*)Bl[buf][nB + 3])[k] = l3;
  };

  v8f acc0 = {}; v8f acc1 = {};

  // ---- software pipeline: prefetch step 0 ----
  float4 a0, a1, bb0, bb1, a0n, a1n, bb0n, bb1n;
  loadA(0, a0, a1);
  loadB(0, bb0, bb1);
  int buf = 0;

  for (int k0 = 0; k0 < KDIM; k0 += 32) {
    // stage current step into LDS[buf]
    stageA(buf, rA,      a0);
    stageA(buf, rA + 32, a1);
    stageB(buf, kB,      bb0);
    stageB(buf, kB + 16, bb1);

    // issue next step's global loads (overlap with barrier + WMMA)
    const bool more = (k0 + 32) < KDIM;
    if (more) { loadA(k0 + 32, a0n, a1n); loadB(k0 + 32, bb0n, bb1n); }

    __syncthreads();

    // ---- fragments per documented 16-bit WMMA VGPR layouts ----
    union Frag { unsigned u[8]; v16bf v; };
    Frag ah, al, bh0, bl0, bh1, bl1;
    const int ma = mrow * 16 + l15;
    #pragma unroll
    for (int r = 0; r < 8; ++r) {
      // A 16x32: VGPR r holds K = 2r + 8g (+8 extra for r>=4), pair-packed
      int kb = 2 * r + 8 * g + ((r >= 4) ? 8 : 0);
      ah.u[r] = Ah[buf][ma][kb >> 1];
      al.u[r] = Al[buf][ma][kb >> 1];
    }
    const int na0 = nc0 + l15;
    const int na1 = na0 + 16;
    #pragma unroll
    for (int r = 0; r < 8; ++r) {
      // B 32x16: VGPR r holds K = 2r + 8g (+8 extra for r>=4), pair-packed, N = lane
      int kb = 2 * r + 8 * g + ((r >= 4) ? 8 : 0);
      bh0.u[r] = Bh[buf][na0][kb >> 1];
      bl0.u[r] = Bl[buf][na0][kb >> 1];
      bh1.u[r] = Bh[buf][na1][kb >> 1];
      bl1.u[r] = Bl[buf][na1][kb >> 1];
    }

    // split-precision MAC: hi*hi + hi*lo + lo*hi (f32 accumulate)
    acc0 = wmma_bf16(ah.v, bh0.v, acc0);
    acc0 = wmma_bf16(ah.v, bl0.v, acc0);
    acc0 = wmma_bf16(al.v, bh0.v, acc0);
    acc1 = wmma_bf16(ah.v, bh1.v, acc1);
    acc1 = wmma_bf16(ah.v, bl1.v, acc1);
    acc1 = wmma_bf16(al.v, bh1.v, acc1);

    a0 = a0n; a1 = a1n; bb0 = bb0n; bb1 = bb1n;
    buf ^= 1;
    // no trailing barrier: with double buffering, the next store targets the
    // other buffer; stores that re-target this buffer occur only after the
    // NEXT iteration's barrier, by which point all reads here are complete.
  }

  // ---- epilogue: bias (+ELU), store per C/D layout (m = 8g+r, n = l15) ----
  const int nA = n0 + nc0 + l15;
  const int nBo = nA + 16;
  const float biasA = (nA  < NDIM) ? be[nA]  : 0.f;
  const float biasB = (nBo < NDIM) ? be[nBo] : 0.f;
  #pragma unroll
  for (int r = 0; r < 8; ++r) {
    int m = m0 + mrow * 16 + g * 8 + r;
    if (m < cnt) {
      float f0 = acc0[r] + biasA;
      float f1 = acc1[r] + biasB;
      if (DO_ELU) {
        f0 = f0 > 0.f ? f0 : expm1f(f0);
        f1 = f1 > 0.f ? f1 : expm1f(f1);
      }
      if (OUT_MODE == 0) {
        outp[(size_t)(off + m) * NDIM + nA]  = f0;
        outp[(size_t)(off + m) * NDIM + nBo] = f1;
      } else {
        int b = rows[off + m];
        if (nA  < NDIM) outp[(size_t)b * NDIM + nA]  = f0;
        if (nBo < NDIM) outp[(size_t)b * NDIM + nBo] = f1;
      }
    }
  }
}

// ------------------------------- launcher ----------------------------------

extern "C" void kernel_launch(void* const* d_in, const int* in_sizes, int n_in,
                              void* d_out, int out_size, void* d_ws, size_t ws_size,
                              hipStream_t stream) {
  const float* x_s   = (const float*)d_in[0];
  const float* x_p   = (const float*)d_in[1];
  const float* W1    = (const float*)d_in[2];
  const float* b1    = (const float*)d_in[3];
  const float* W2    = (const float*)d_in[4];
  const float* b2    = (const float*)d_in[5];
  const float* W3    = (const float*)d_in[6];
  const float* b3    = (const float*)d_in[7];
  const int*   label = (const int*)d_in[8];
  float* out = (float*)d_out;

  int* hdr  = (int*)d_ws;                                  // 32 ints
  int* rows = hdr + 32;                                    // B_ ints
  float* h1 = (float*)((char*)d_ws + 65536);               // [B_, H1_] f32 (16 MB)
  float* h2 = h1 + (size_t)B_ * H1_;                       // [B_, H2_] f32 ( 8 MB)

  k_zero_hdr<<<1, 32, 0, stream>>>(hdr);
  k_count<<<(B_ + 255) / 256, 256, 0, stream>>>(label, hdr);
  k_scan<<<1, 1, 0, stream>>>(hdr);
  k_fill<<<(B_ + 255) / 256, 256, 0, stream>>>(label, hdr, rows);

  // L1: [cnt_e, 1024] x [1024, 1024] + bias, ELU   (x = concat(x_p, x_s))
  moe_mlp_layer<2 * L_, H1_, true, 0, 0>
      <<<dim3(B_ / 64, H1_ / 64, E_), 256, 0, stream>>>(x_p, x_s, W1, b1, hdr, rows, h1);
  // L2: [cnt_e, 1024] x [1024, 512] + bias, ELU
  moe_mlp_layer<H1_, H2_, true, 1, 0>
      <<<dim3(B_ / 64, H2_ / 64, E_), 256, 0, stream>>>(h1, nullptr, W2, b2, hdr, rows, h2);
  // L3: [cnt_e, 512] x [512, 40] + bias, scatter to d_out[b]
  moe_mlp_layer<H2_, C_, false, 1, 1>
      <<<dim3(B_ / 64, 1, E_), 256, 0, stream>>>(h2, nullptr, W3, b3, hdr, rows, out);
}
